// TextProposalWithAttn_2542620639837
// MI455X (gfx1250) — compile-verified, matrix operand order checked by reading, not run on hardware
//
#include <hip/hip_runtime.h>
#include <hip/hip_bf16.h>

// ---------------------------------------------------------------------------
// TextProposalWithAttn forward for MI455X (gfx1250, wave32).
// GEMMs use v_wmma_f32_16x16x32_bf16 (f32 accumulate) with 4-way M register
// blocking: each wave owns a full 64-row column (4 accumulator tiles), so the
// streaming weight operand (W_out, L2-resident: 32 MB bf16 << 192 MB L2) is
// read once per N-tile instead of 4x. This keeps the net bound by the
// irreducible 262 MB f32 output write stream (~13 us at 23.3 TB/s).
// ---------------------------------------------------------------------------

#define B_   64
#define S_   49
#define T_   32
#define ENC_ 512
#define HID_ 512
#define EMB_ 512
#define VOC_ 32000

typedef __bf16 v16bf __attribute__((ext_vector_type(16)));
typedef float  v8f   __attribute__((ext_vector_type(8)));

union FragBF { v16bf v; uint4 q[2]; };

__device__ __forceinline__ float sigmoidf_(float x) {
    return 1.0f / (1.0f + __expf(-x));
}

// ---------------------------------------------------------------------------
// f32 -> bf16 conversion (grid-stride)
// ---------------------------------------------------------------------------
__global__ void cvt_f32_bf16(const float* __restrict__ src,
                             __bf16* __restrict__ dst, long n) {
    long i = (long)blockIdx.x * blockDim.x + threadIdx.x;
    long stride = (long)gridDim.x * blockDim.x;
    for (; i < n; i += stride) dst[i] = (__bf16)src[i];
}

// ---------------------------------------------------------------------------
// WMMA GEMM with 4-way M register blocking:
//   C[M,N] = A[M,K] @ W[N,K]^T (+bias[n]) (* act-mask[m])
// A, W bf16 row-major (K contiguous), C f32 with row stride ldc.
// Each wave: one 16-col N-tile x 64 rows (4 accumulator tiles).
// grid = (ceil(N/16 / wavesPerBlock), M/64). M multiple of 64, K of 32.
// ---------------------------------------------------------------------------
__global__ void wmma_gemm_bias(const __bf16* __restrict__ A,
                               const __bf16* __restrict__ W,
                               const float* __restrict__ bias,
                               float* __restrict__ C,
                               int M, int N, int K, long ldc,
                               const int* __restrict__ length, int tstep) {
    const int lane  = threadIdx.x & 31;
    const int wave  = threadIdx.x >> 5;
    const int ntile = blockIdx.x * (blockDim.x >> 5) + wave;
    const int n0 = ntile * 16;
    const int m0 = blockIdx.y * 64;
    if (n0 >= N) return;

    const int half = lane >> 4;   // 0: K offset 0..; 1: K offset +8
    const int lm   = lane & 15;

    const __bf16* wrow = W + (long)(n0 + lm) * K + 8 * half;
    const __bf16* arow = A + (long)(m0 + lm) * K + 8 * half;
    const long mstride = (long)16 * K;   // lane's next m-tile row, in elements

    v8f acc[4] = {};
    for (int k0 = 0; k0 < K; k0 += 32) {
        FragBF b;
        b.q[0] = *(const uint4*)(wrow + k0);
        b.q[1] = *(const uint4*)(wrow + k0 + 16);
        // prefetch the streaming W operand ahead (global_prefetch_b8)
        __builtin_prefetch(wrow + k0 + 64, 0, 1);

        FragBF a[4];
#pragma unroll
        for (int mt = 0; mt < 4; ++mt) {
            a[mt].q[0] = *(const uint4*)(arow + (long)mt * mstride + k0);
            a[mt].q[1] = *(const uint4*)(arow + (long)mt * mstride + k0 + 16);
        }
#pragma unroll
        for (int mt = 0; mt < 4; ++mt) {
            acc[mt] = __builtin_amdgcn_wmma_f32_16x16x32_bf16(
                          false, a[mt].v, false, b.v, (short)0, acc[mt],
                          false, false);
        }
    }

    const int n = n0 + lm;
    const float bv = bias ? bias[n] : 0.0f;
#pragma unroll
    for (int mt = 0; mt < 4; ++mt) {
#pragma unroll
        for (int v = 0; v < 8; ++v) {
            const int m = m0 + mt * 16 + v + 8 * half;
            float val = acc[mt][v] + bv;
            if (length) val *= (tstep < length[m]) ? 1.0f : 0.0f;
            C[(long)m * ldc + n] = val;
        }
    }
}

// ---------------------------------------------------------------------------
// h0 = mean_s(feats) @ W_init^T + b_init ; scope = 1
// one block per batch row
// ---------------------------------------------------------------------------
__global__ void h0_kernel(const float* __restrict__ feats,
                          const float* __restrict__ W_init,
                          const float* __restrict__ b_init,
                          float* __restrict__ h,
                          float* __restrict__ scope) {
    const int b = blockIdx.x;
    const int tid = threadIdx.x;
    __shared__ float mf[ENC_];

    for (int e = tid; e < ENC_; e += blockDim.x) {
        float s = 0.0f;
        const float* f = feats + ((long)b * S_) * ENC_ + e;
        for (int ss = 0; ss < S_; ++ss) s += f[(long)ss * ENC_];
        mf[e] = s * (1.0f / (float)S_);
    }
    __syncthreads();

    for (int o = tid; o < HID_; o += blockDim.x) {
        float d = b_init[o];
        const float* row = W_init + (long)o * ENC_;
#pragma unroll 4
        for (int k = 0; k < ENC_; ++k) d += mf[k] * row[k];
        h[(long)b * HID_ + o] = d;
    }
    if (tid < S_) scope[b * S_ + tid] = 1.0f;
}

// ---------------------------------------------------------------------------
// Fused per-step attention: att2, relu-dot scores, softmax, alpha_tilde,
// scope update, awe, gate, and assembly of bf16 GRU input x=[emb, gate*awe].
// One block (256 threads = 8 waves) per batch row.
// ---------------------------------------------------------------------------
__global__ void attn_fused(const float* __restrict__ feats,
                           const float* __restrict__ att1,
                           const float* __restrict__ W_da, const float* __restrict__ b_da,
                           const float* __restrict__ W_beta, const float* __restrict__ b_beta,
                           const float* __restrict__ W_fa, const float* __restrict__ b_fa,
                           const float* __restrict__ emb,
                           const int* __restrict__ seq,
                           const int* __restrict__ length,
                           float* __restrict__ h,
                           float* __restrict__ scope,
                           __bf16* __restrict__ h_bf,
                           __bf16* __restrict__ x_bf,
                           float* __restrict__ alphas_out,
                           int tstep) {
    const int b = blockIdx.x;
    const int tid = threadIdx.x;
    const int lane = tid & 31;
    const int wv = tid >> 5;

    __shared__ float h_s[HID_];
    __shared__ float att2_s[HID_];
    __shared__ float e_s[S_];
    __shared__ float alpha_s[S_];

    const int act = (tstep < length[b]) ? 1 : 0;
    const float actf = act ? 1.0f : 0.0f;

    // load h into LDS, also produce bf16 copy of h for the W_hh GEMM
    for (int e = tid; e < HID_; e += blockDim.x) {
        float hv = h[(long)b * HID_ + e];
        h_s[e] = hv;
        h_bf[(long)b * HID_ + e] = (__bf16)hv;
    }
    __syncthreads();

    // att2 = h @ W_da^T + b_da
    for (int a = tid; a < HID_; a += blockDim.x) {
        float d = b_da[a];
        const float* row = W_da + (long)a * HID_;
#pragma unroll 4
        for (int k = 0; k < HID_; ++k) d += h_s[k] * row[k];
        att2_s[a] = d;
    }
    __syncthreads();

    // e[s] = sum_a relu(att1 + att2) * W_fa[a] + b_fa   (one wave per s)
    for (int s = wv; s < S_; s += (blockDim.x >> 5)) {
        const float* a1 = att1 + ((long)b * S_ + s) * HID_;
        float p = 0.0f;
        for (int a = lane; a < HID_; a += 32) {
            float u = a1[a] + att2_s[a];
            u = fmaxf(u, 0.0f);
            p += u * W_fa[a];
        }
#pragma unroll
        for (int off = 16; off > 0; off >>= 1) p += __shfl_down(p, off, 32);
        if (lane == 0) e_s[s] = p + b_fa[0];
    }
    __syncthreads();

    // softmax over S (tiny: serial on one thread)
    if (tid == 0) {
        float mx = e_s[0];
        for (int s = 1; s < S_; ++s) mx = fmaxf(mx, e_s[s]);
        float sum = 0.0f;
        for (int s = 0; s < S_; ++s) { float e = __expf(e_s[s] - mx); alpha_s[s] = e; sum += e; }
        float inv = 1.0f / sum;
        for (int s = 0; s < S_; ++s) alpha_s[s] *= inv;
    }
    __syncthreads();

    // alpha_tilde output + scope update
    if (tid < S_) {
        float al = alpha_s[tid];
        float sc = scope[b * S_ + tid];
        alphas_out[((long)b * T_ + tstep) * S_ + tid] = sc * al * actf;
        if (act) scope[b * S_ + tid] = sc * (1.0f - al);
    }

    // awe, gate, and x = [emb[token], gate*awe] in bf16
    const int tok = seq[b * T_ + tstep];
    for (int e = tid; e < ENC_; e += blockDim.x) {
        float awe = 0.0f;
        const float* f = feats + ((long)b * S_) * ENC_ + e;
        for (int ss = 0; ss < S_; ++ss) awe += f[(long)ss * ENC_] * alpha_s[ss];

        float g = b_beta[e];
        const float* row = W_beta + (long)e * HID_;
#pragma unroll 4
        for (int k = 0; k < HID_; ++k) g += h_s[k] * row[k];
        g = sigmoidf_(g);

        x_bf[(long)b * (EMB_ + ENC_) + e]        = (__bf16)emb[(long)tok * EMB_ + e];
        x_bf[(long)b * (EMB_ + ENC_) + EMB_ + e] = (__bf16)(g * awe);
    }
}

// ---------------------------------------------------------------------------
// GRU elementwise update: r,z,n gates, h_new; h = act ? h_new : h;
// h_new (unmasked) stored as bf16 for the vocab projection.
// ---------------------------------------------------------------------------
__global__ void gru_update(const float* __restrict__ gi,
                           const float* __restrict__ gh,
                           float* __restrict__ h,
                           const int* __restrict__ length,
                           __bf16* __restrict__ hn_bf,
                           int tstep) {
    const int idx = blockIdx.x * blockDim.x + threadIdx.x;
    if (idx >= B_ * HID_) return;
    const int b = idx >> 9;      // /512
    const int e = idx & 511;

    const float* gib = gi + (long)b * (3 * HID_);
    const float* ghb = gh + (long)b * (3 * HID_);
    const float ho = h[idx];

    const float r = sigmoidf_(gib[e]            + ghb[e]);
    const float z = sigmoidf_(gib[HID_ + e]     + ghb[HID_ + e]);
    const float n = tanhf   (gib[2 * HID_ + e] + r * ghb[2 * HID_ + e]);
    const float hn = (1.0f - z) * n + z * ho;

    const int act = (tstep < length[b]) ? 1 : 0;
    h[idx] = act ? hn : ho;
    hn_bf[idx] = (__bf16)hn;
}

// ---------------------------------------------------------------------------
// host-side launch
// ---------------------------------------------------------------------------
static inline char* carve(char*& p, size_t bytes) {
    char* r = p;
    p += (bytes + 255) & ~(size_t)255;
    return r;
}

extern "C" void kernel_launch(void* const* d_in, const int* in_sizes, int n_in,
                              void* d_out, int out_size, void* d_ws, size_t ws_size,
                              hipStream_t stream) {
    const float* feats  = (const float*)d_in[0];
    const int*   seq    = (const int*)  d_in[1];
    const int*   length = (const int*)  d_in[2];
    const float* emb    = (const float*)d_in[3];
    const float* W_ih   = (const float*)d_in[4];
    const float* b_ih   = (const float*)d_in[5];
    const float* W_hh   = (const float*)d_in[6];
    const float* b_hh   = (const float*)d_in[7];
    const float* W_out  = (const float*)d_in[8];
    const float* b_out  = (const float*)d_in[9];
    const float* W_init = (const float*)d_in[10];
    const float* b_init = (const float*)d_in[11];
    const float* W_beta = (const float*)d_in[12];
    const float* b_beta = (const float*)d_in[13];
    const float* W_ea   = (const float*)d_in[14];
    const float* b_ea   = (const float*)d_in[15];
    const float* W_da   = (const float*)d_in[16];
    const float* b_da   = (const float*)d_in[17];
    const float* W_fa   = (const float*)d_in[18];
    const float* b_fa   = (const float*)d_in[19];

    char* p = (char*)d_ws;
    __bf16* W_out_bf = (__bf16*)carve(p, (size_t)VOC_ * HID_ * 2);
    __bf16* W_ih_bf  = (__bf16*)carve(p, (size_t)3 * HID_ * (EMB_ + ENC_) * 2);
    __bf16* W_hh_bf  = (__bf16*)carve(p, (size_t)3 * HID_ * HID_ * 2);
    __bf16* W_ea_bf  = (__bf16*)carve(p, (size_t)HID_ * ENC_ * 2);
    __bf16* feats_bf = (__bf16*)carve(p, (size_t)B_ * S_ * ENC_ * 2);
    float*  att1     = (float*) carve(p, (size_t)B_ * S_ * HID_ * 4);
    float*  h        = (float*) carve(p, (size_t)B_ * HID_ * 4);
    __bf16* h_bf     = (__bf16*)carve(p, (size_t)B_ * HID_ * 2);
    __bf16* hn_bf    = (__bf16*)carve(p, (size_t)B_ * HID_ * 2);
    __bf16* x_bf     = (__bf16*)carve(p, (size_t)B_ * (EMB_ + ENC_) * 2);
    float*  gi       = (float*) carve(p, (size_t)B_ * 3 * HID_ * 4);
    float*  gh       = (float*) carve(p, (size_t)B_ * 3 * HID_ * 4);
    float*  scope    = (float*) carve(p, (size_t)B_ * S_ * 4);

    float* out_f  = (float*)d_out;                       // [B, T, VOCAB]
    float* alphas = out_f + (long)B_ * T_ * VOC_;        // [B, T, S]

    // one-time weight / activation conversions to bf16
    cvt_f32_bf16<<<2048, 256, 0, stream>>>(W_out, W_out_bf, (long)VOC_ * HID_);
    cvt_f32_bf16<<<1024, 256, 0, stream>>>(W_ih,  W_ih_bf,  (long)3 * HID_ * (EMB_ + ENC_));
    cvt_f32_bf16<<<1024, 256, 0, stream>>>(W_hh,  W_hh_bf,  (long)3 * HID_ * HID_);
    cvt_f32_bf16<<<512,  256, 0, stream>>>(W_ea,  W_ea_bf,  (long)HID_ * ENC_);
    cvt_f32_bf16<<<1024, 256, 0, stream>>>(feats, feats_bf, (long)B_ * S_ * ENC_);

    // h0 + scope init
    h0_kernel<<<B_, 256, 0, stream>>>(feats, W_init, b_init, h, scope);

    // att1 = feats @ W_ea^T + b_ea : [3136, 512] = [3136,512] @ [512,512]^T
    // grid: (Ntiles/4 waves, M/64)
    wmma_gemm_bias<<<dim3((HID_ / 16) / 4, (B_ * S_) / 64), 128, 0, stream>>>(
        feats_bf, W_ea_bf, b_ea, att1, B_ * S_, HID_, ENC_, HID_, nullptr, 0);

    for (int t = 0; t < T_; ++t) {
        attn_fused<<<B_, 256, 0, stream>>>(feats, att1, W_da, b_da, W_beta, b_beta,
                                           W_fa, b_fa, emb, seq, length,
                                           h, scope, h_bf, x_bf, alphas, t);

        // gi = x @ W_ih^T + b_ih : [64,1536], K=1024
        wmma_gemm_bias<<<dim3((3 * HID_ / 16) / 4, 1), 128, 0, stream>>>(
            x_bf, W_ih_bf, b_ih, gi, B_, 3 * HID_, EMB_ + ENC_, 3 * HID_, nullptr, 0);

        // gh = h @ W_hh^T + b_hh : [64,1536], K=512
        wmma_gemm_bias<<<dim3((3 * HID_ / 16) / 4, 1), 128, 0, stream>>>(
            h_bf, W_hh_bf, b_hh, gh, B_, 3 * HID_, HID_, 3 * HID_, nullptr, 0);

        gru_update<<<(B_ * HID_ + 255) / 256, 256, 0, stream>>>(gi, gh, h, length, hn_bf, t);

        // preds = h_new @ W_out^T + b_out, masked by active[t,b]
        // C row b -> out_f[b*T*VOC + t*VOC + n]
        wmma_gemm_bias<<<dim3((VOC_ / 16) / 4, 1), 128, 0, stream>>>(
            hn_bf, W_out_bf, b_out, out_f + (long)t * VOC_,
            B_, VOC_, HID_, (long)T_ * VOC_, length, t);
    }
}
